// FidelityLSTM_62861141344460
// MI455X (gfx1250) — matrix operand order repaired and checked by reading, not compiled, not run on hardware
//
#include <hip/hip_runtime.h>
#include <hip/hip_bf16.h>
#include <math.h>

// Problem constants (match reference)
#define BB   256
#define TT   256
#define II   512
#define HH   1024
#define G4H  4096   // 4*H
#define BT   (BB*TT)

typedef __attribute__((ext_vector_type(16))) __bf16 v16bf;
typedef __attribute__((ext_vector_type(8)))  __bf16 v8bf;
typedef __attribute__((ext_vector_type(8)))  float  v8f;

// ---------------------------------------------------------------------------
// Elementwise prep kernels
// ---------------------------------------------------------------------------
__global__ void cvt_f32_to_bf16_kernel(const float* __restrict__ in,
                                       __bf16* __restrict__ out, int n) {
  int i = blockIdx.x * blockDim.x + threadIdx.x;
  if (i < n) out[i] = (__bf16)in[i];
}

__global__ void zero_u32_kernel(unsigned* __restrict__ p, int n) {
  int i = blockIdx.x * blockDim.x + threadIdx.x;
  if (i < n) p[i] = 0u;
}

// Pack W [N,K] (row-major f32) into WMMA B-operand tiles of bf16.
// Tile = 32(K) x 16(N); tileIdx = ntile*(K/32) + ktile; 512 bf16 per tile.
// Within tile: flat index = lane*16 + e, with element e of lane l holding
//   B[k,n] where k = ktile*32 + (e&7) + (e>=8 ? 16 : 0) + (l>=16 ? 8 : 0), n = ntile*16 + (l&15),
// and B[k,n] = W[n,k]  (B is W transposed: C = A * W^T).
__global__ void pack_w_kernel(const float* __restrict__ W,
                              __bf16* __restrict__ out, int N, int K) {
  int i = blockIdx.x * blockDim.x + threadIdx.x;
  if (i >= N * K) return;
  int tile = i >> 9;          // /512
  int s    = i & 511;
  int lane = s >> 4;
  int e    = s & 15;
  int KT   = K >> 5;
  int ntile = tile / KT;
  int ktile = tile - ntile * KT;
  int k = (ktile << 5) + (e & 7) + ((e >> 3) << 4) + ((lane >= 16) ? 8 : 0);
  int n = (ntile << 4) + (lane & 15);
  out[i] = (__bf16)W[(size_t)n * K + k];
}

// ---------------------------------------------------------------------------
// A-operand loader per CDNA5 WMMA 16-bit A layout (16x32, MxK):
// lane l (row M = l&15): elements 0..7 = K kb+off..kb+off+7,
//                        elements 8..15 = K kb+off+16..kb+off+23,
// with off = (l>=16) ? 8 : 0  (folded into arow).
// ---------------------------------------------------------------------------
__device__ __forceinline__ v16bf load_a_tile(const __bf16* __restrict__ arow,
                                             int kb) {
  v8bf lo = *(const v8bf*)(arow + kb);
  v8bf hi = *(const v8bf*)(arow + kb + 16);
  return __builtin_shufflevector(lo, hi, 0, 1, 2, 3, 4, 5, 6, 7,
                                 8, 9, 10, 11, 12, 13, 14, 15);
}

// ---------------------------------------------------------------------------
// Async global -> LDS copy of 16 bytes per lane (CDNA5, ASYNCcnt-tracked).
// lds_off is the wave-relative LDS byte address (low 32 bits of a generic
// pointer to a __shared__ object, per ISA flat->LDS aperture truncation).
// ---------------------------------------------------------------------------
__device__ __forceinline__ void async_copy_b128(unsigned lds_off,
                                                const void* gptr) {
  asm volatile("global_load_async_to_lds_b128 %0, %1, off"
               :: "v"(lds_off), "v"(gptr)
               : "memory");
}
__device__ __forceinline__ void wait_asynccnt0() {
  asm volatile("s_wait_asynccnt 0x0" ::: "memory");
}

// ---------------------------------------------------------------------------
// Device-wide sense-reversing barrier (persistent-kernel timestep sync).
// Representative-thread fence + workgroup barrier pattern; gen is read
// BEFORE arrival so late spinners can never miss a generation bump.
// ---------------------------------------------------------------------------
__device__ __forceinline__ void grid_barrier(unsigned* cnt, unsigned* gen,
                                             unsigned nblk) {
  __syncthreads();
  if (threadIdx.x == 0) {
    unsigned g = atomicAdd(gen, 0u);
    __threadfence();                       // release h-writes device-wide
    if (atomicAdd(cnt, 1u) == nblk - 1) {
      atomicExch(cnt, 0u);                 // reset for next round
      atomicAdd(gen, 1u);                  // open the barrier
    } else {
      while (atomicAdd(gen, 0u) == g) __builtin_amdgcn_s_sleep(2);
    }
    __threadfence();                       // acquire others' h-writes
  }
  __syncthreads();
}

// ---------------------------------------------------------------------------
// Input-contribution GEMM: C[M,N] = A[M,K](bf16) * W^T (packed bf16)
//                                   + bias_a[N] + bias_b[N]   (f32 out)
// Block: 256 threads = 8 waves; wave w -> 32 rows x 64 cols (2 M-tiles x
// 4 N-tiles = 8 accumulators, B reused across M-tiles). Block covers
// 256 rows x 64 cols. grid = (N/64, M/256).
// Copy-free ping-pong software pipeline over two register sets (KT even).
// ---------------------------------------------------------------------------
__global__ __launch_bounds__(256)
void gemm_bf16_wmma_kernel(const __bf16* __restrict__ A,
                           const __bf16* __restrict__ Wp,
                           const float* __restrict__ bias_a,
                           const float* __restrict__ bias_b,
                           float* __restrict__ C, int M, int N, int K) {
  const int lane  = threadIdx.x & 31;
  const int w     = threadIdx.x >> 5;
  const int mbase = blockIdx.y * 256 + w * 32;
  const int n0    = blockIdx.x * 64;
  const int KT    = K >> 5;
  const int ntb   = blockIdx.x * 4;

  v8f acc[2][4] = {};
  const __bf16* arow0 = A + (size_t)(mbase + (lane & 15)) * K + ((lane >> 4) << 3);
  const __bf16* arow1 = arow0 + (size_t)16 * K;
  const __bf16* bp    = Wp + (size_t)lane * 16;

  v16bf aP[2], aQ[2], bP[4], bQ[4];

  auto loadA = [&](v16bf (&a)[2], int kt) {
    a[0] = load_a_tile(arow0, kt * 32);
    a[1] = load_a_tile(arow1, kt * 32);
  };
  auto loadB = [&](v16bf (&b)[4], int kt) {
#pragma unroll
    for (int q = 0; q < 4; ++q)
      b[q] = *(const v16bf*)(bp + ((size_t)(ntb + q) * KT + kt) * 512);
  };
  auto mma = [&](v16bf (&a)[2], v16bf (&b)[4]) {
#pragma unroll
    for (int m = 0; m < 2; ++m)
#pragma unroll
      for (int q = 0; q < 4; ++q)
        acc[m][q] = __builtin_amdgcn_wmma_f32_16x16x32_bf16(
            false, a[m], false, b[q], (short)0, acc[m][q], false, false);
  };

  // Pipeline: P holds k-tile kt, Q holds kt+1; no rotations, no tail branch.
  loadA(aP, 0);
  loadB(bP, 0);
#pragma unroll 1
  for (int kt = 0; kt + 2 < KT; kt += 2) {
    loadA(aQ, kt + 1);
    loadB(bQ, kt + 1);
    mma(aP, bP);
    loadA(aP, kt + 2);
    loadB(bP, kt + 2);
    mma(aQ, bQ);
  }
  loadA(aQ, KT - 1);
  loadB(bQ, KT - 1);
  mma(aP, bP);
  mma(aQ, bQ);

  // C/D layout: VGPR r, lanes 0-15 -> M=r, lanes 16-31 -> M=r+8; N = lane&15
  const int ml = (lane >> 4) * 8;
  const int nl = lane & 15;
#pragma unroll
  for (int m = 0; m < 2; ++m) {
#pragma unroll
    for (int q = 0; q < 4; ++q) {
      const int ncol = n0 + q * 16 + nl;
      const float bsum = bias_a[ncol] + bias_b[ncol];
#pragma unroll
      for (int r = 0; r < 8; ++r) {
        const int row = mbase + m * 16 + ml + r;
        C[(size_t)row * N + ncol] = acc[m][q][r] + bsum;
      }
    }
  }
}

// ---------------------------------------------------------------------------
// Persistent LSTM sequence kernel (one launch per layer, all T timesteps).
//
//   gates[B,4H] = xg[:,t,:] + h_{t-1} @ W_hh^T ; LSTM cell ; h_t -> hall
//
// - Each block owns a 16-wide hidden-unit slice (jt) for 128 batch rows
//   (8 waves x 16). Its W_hh panel (4 gates x 16 cols x K=1024 = 128 KB
//   bf16) is staged into LDS ONCE with global_load_async_to_lds_b128 and
//   reused for all T steps: steady state reads weights only via
//   ds_load_b128 (CDNA5 WGP LDS = 320 KB; 256 thr + 128 KB -> 2 blocks/WGP).
// - Cell state c lives in registers for the whole sequence (8 f32/lane).
// - Each wave accumulates all 4 gates for its 16x16 cell tile (4 WMMA
//   accumulators sharing one A operand) -> elementwise cell needs no
//   cross-wave exchange.
// - A operand (h_{t-1}, global, written by all blocks in step t-1) is
//   register ping-ponged across k-tiles; timesteps are separated by a
//   device-wide barrier.
//
// grid = (H/16, B/128) = (64, 2) = 128 blocks, 256 threads.
// ---------------------------------------------------------------------------
#define STEP_KT (HH >> 5)   // 32 k-tiles (even)

__global__ __launch_bounds__(256)
void lstm_seq_kernel(const __bf16* __restrict__ h0,     // [B,H] initial hidden
                     const __bf16* __restrict__ Whhp,   // packed W_hh
                     const float*  __restrict__ xg,     // [B*T, 4H]
                     __bf16* __restrict__ hall,         // [B*T, H] hidden out
                     unsigned* __restrict__ bar_cnt,
                     unsigned* __restrict__ bar_gen) {
  extern __shared__ __bf16 lds_w[];    // [4][STEP_KT][512] bf16 = 128 KB

  const int tid   = threadIdx.x;
  const int lane  = tid & 31;
  const int w     = tid >> 5;
  const int jt    = blockIdx.x;                    // hidden-unit tile
  const int j0    = jt * 16;
  const int bbase = (blockIdx.y * 8 + w) * 16;     // wave's batch rows
  const unsigned nblk = gridDim.x * gridDim.y;

  // ---- one-time async stage of this block's weight panel into LDS ----
  const unsigned lds0 = (unsigned)(uintptr_t)&lds_w[0];
  for (int c = tid; c < 8192; c += 256) {          // 8192 x 16B = 128 KB
    const int g = c >> 11;                         // 2048 chunks per gate
    const int r = c & 2047;
    const __bf16* src =
        Whhp + ((size_t)(g * (HH / 16) + jt) * STEP_KT) * 512 + (size_t)r * 8;
    async_copy_b128(lds0 + (unsigned)c * 16, src);
  }
  wait_asynccnt0();
  __syncthreads();

  const int ml    = (lane >> 4) * 8;
  const int nl    = lane & 15;
  const int j     = j0 + nl;
  const int koff  = (lane >> 4) << 3;
  const int arowr = bbase + (lane & 15);

  float cst[8] = {0.f, 0.f, 0.f, 0.f, 0.f, 0.f, 0.f, 0.f};

  for (int t = 0; t < TT; ++t) {
    const __bf16* arow =
        (t == 0) ? (h0 + (size_t)arowr * HH + koff)
                 : (hall + ((size_t)arowr * TT + (t - 1)) * (size_t)HH + koff);

    v8f acc[4] = {};
    auto mma_lds = [&](int kt, v16bf a) {
      v16bf b[4];
#pragma unroll
      for (int g = 0; g < 4; ++g)
        b[g] = *(const v16bf*)(&lds_w[((size_t)g * STEP_KT + kt) * 512 + lane * 16]);
#pragma unroll
      for (int g = 0; g < 4; ++g)
        acc[g] = __builtin_amdgcn_wmma_f32_16x16x32_bf16(
            false, a, false, b[g], (short)0, acc[g], false, false);
    };

    // copy-free A ping-pong; final kt+2 load is a harmless in-workspace
    // overfetch (never consumed).
    v16bf aP = load_a_tile(arow, 0), aQ;
#pragma unroll 1
    for (int kt = 0; kt < STEP_KT; kt += 2) {
      aQ = load_a_tile(arow, (kt + 1) * 32);
      mma_lds(kt, aP);
      aP = load_a_tile(arow, (kt + 2) * 32);
      mma_lds(kt + 1, aQ);
    }

    // ---- elementwise LSTM cell; c state stays in registers ----
#pragma unroll
    for (int r = 0; r < 8; ++r) {
      const int b = bbase + ml + r;
      const size_t xrow = ((size_t)b * TT + t) * (size_t)G4H;
      float gi = acc[0][r] + xg[xrow + j];
      float gf = acc[1][r] + xg[xrow + HH + j];
      float gg = acc[2][r] + xg[xrow + 2 * HH + j];
      float go = acc[3][r] + xg[xrow + 3 * HH + j];
      gi = 1.0f / (1.0f + __expf(-gi));
      gf = 1.0f / (1.0f + __expf(-gf));
      gg = tanhf(gg);
      go = 1.0f / (1.0f + __expf(-go));
      cst[r] = gf * cst[r] + gi * gg;
      const float h = go * tanhf(cst[r]);
      hall[((size_t)b * TT + t) * (size_t)HH + j] = (__bf16)h;
    }

    grid_barrier(bar_cnt, bar_gen, nblk);
  }
}

// ---------------------------------------------------------------------------
// Readout: out[b] = h2[b, T-1, :] . W_ro + b_ro   (O = 1)
// ---------------------------------------------------------------------------
__global__ __launch_bounds__(256)
void readout_kernel(const __bf16* __restrict__ h2,
                    const float* __restrict__ Wro,
                    const float* __restrict__ bro,
                    float* __restrict__ out) {
  __shared__ float red[256];
  const int b = blockIdx.x;
  const __bf16* hrow = h2 + ((size_t)b * TT + (TT - 1)) * (size_t)HH;
  float s = 0.0f;
  for (int jj = threadIdx.x; jj < HH; jj += 256)
    s += (float)hrow[jj] * Wro[jj];
  red[threadIdx.x] = s;
  __syncthreads();
  for (int off = 128; off > 0; off >>= 1) {
    if (threadIdx.x < off) red[threadIdx.x] += red[threadIdx.x + off];
    __syncthreads();
  }
  if (threadIdx.x == 0) out[b] = red[0] + bro[0];
}

// ---------------------------------------------------------------------------
// Host-side orchestration
// ---------------------------------------------------------------------------
extern "C" void kernel_launch(void* const* d_in, const int* in_sizes, int n_in,
                              void* d_out, int out_size, void* d_ws, size_t ws_size,
                              hipStream_t stream) {
  (void)in_sizes; (void)n_in; (void)out_size; (void)ws_size;

  const float* init_hidden = (const float*)d_in[0];   // [B,H]
  const float* context     = (const float*)d_in[1];   // [B,T,I]
  const float* W_ih0       = (const float*)d_in[2];   // [4H,I]
  const float* W_hh0       = (const float*)d_in[3];   // [4H,H]
  const float* b_ih0       = (const float*)d_in[4];
  const float* b_hh0       = (const float*)d_in[5];
  const float* W_ih1       = (const float*)d_in[6];   // [4H,H]
  const float* W_hh1       = (const float*)d_in[7];   // [4H,H]
  const float* b_ih1       = (const float*)d_in[8];
  const float* b_hh1       = (const float*)d_in[9];
  const float* W_ro        = (const float*)d_in[10];  // [1,H]
  const float* b_ro        = (const float*)d_in[11];  // [1]
  float* out = (float*)d_out;

  // Workspace carve-up (256B aligned slices)
  char* ws = (char*)d_ws;
  size_t off = 0;
  auto carve = [&](size_t bytes) -> void* {
    off = (off + 255) & ~(size_t)255;
    void* p = ws + off;
    off += bytes;
    return p;
  };
  __bf16*   Xbf   = (__bf16*)carve((size_t)BT * II * sizeof(__bf16));     //  64 MB
  __bf16*   Wih0p = (__bf16*)carve((size_t)G4H * II * sizeof(__bf16));    //   4 MB
  __bf16*   Whh0p = (__bf16*)carve((size_t)G4H * HH * sizeof(__bf16));    //   8 MB
  __bf16*   Wih1p = (__bf16*)carve((size_t)G4H * HH * sizeof(__bf16));    //   8 MB
  __bf16*   Whh1p = (__bf16*)carve((size_t)G4H * HH * sizeof(__bf16));    //   8 MB
  __bf16*   h0bf  = (__bf16*)carve((size_t)BB * HH * sizeof(__bf16));     // 0.5 MB
  unsigned* bar   = (unsigned*)carve(256);                                // barrier state
  float*    xgbuf = (float*)carve((size_t)BT * G4H * sizeof(float));      //   1 GB
  __bf16*   h1all = (__bf16*)carve((size_t)BT * HH * sizeof(__bf16));     // 128 MB
  __bf16*   h2all = (__bf16*)carve((size_t)BT * HH * sizeof(__bf16));     // 128 MB

  const size_t LDS_BYTES = 4u * STEP_KT * 512u * sizeof(__bf16);          // 128 KB

  // --- prep: conversions + weight packing ---
  {
    int n = BT * II;
    cvt_f32_to_bf16_kernel<<<(n + 255) / 256, 256, 0, stream>>>(context, Xbf, n);
  }
  {
    int n = BB * HH;
    cvt_f32_to_bf16_kernel<<<(n + 255) / 256, 256, 0, stream>>>(init_hidden, h0bf, n);
  }
  {
    int n = G4H * II;
    pack_w_kernel<<<(n + 255) / 256, 256, 0, stream>>>(W_ih0, Wih0p, G4H, II);
  }
  {
    int n = G4H * HH;
    pack_w_kernel<<<(n + 255) / 256, 256, 0, stream>>>(W_hh0, Whh0p, G4H, HH);
    pack_w_kernel<<<(n + 255) / 256, 256, 0, stream>>>(W_ih1, Wih1p, G4H, HH);
    pack_w_kernel<<<(n + 255) / 256, 256, 0, stream>>>(W_hh1, Whh1p, G4H, HH);
  }

  // ================= Layer 0 =================
  zero_u32_kernel<<<1, 64, 0, stream>>>(bar, 64);
  // xg0 = X @ W_ih0^T + b_ih0 + b_hh0     [BT, 4H]
  gemm_bf16_wmma_kernel<<<dim3(G4H / 64, BT / 256), 256, 0, stream>>>(
      Xbf, Wih0p, b_ih0, b_hh0, xgbuf, BT, G4H, II);
  lstm_seq_kernel<<<dim3(HH / 16, BB / 128), 256, LDS_BYTES, stream>>>(
      h0bf, Whh0p, xgbuf, h1all, bar, bar + 1);

  // ================= Layer 1 =================
  zero_u32_kernel<<<1, 64, 0, stream>>>(bar, 64);
  // xg1 = h1 @ W_ih1^T + b_ih1 + b_hh1    [BT, 4H]
  gemm_bf16_wmma_kernel<<<dim3(G4H / 64, BT / 256), 256, 0, stream>>>(
      h1all, Wih1p, b_ih1, b_hh1, xgbuf, BT, G4H, HH);
  lstm_seq_kernel<<<dim3(HH / 16, BB / 128), 256, LDS_BYTES, stream>>>(
      h0bf, Whh1p, xgbuf, h2all, bar, bar + 1);

  // ================= Readout =================
  readout_kernel<<<BB, 256, 0, stream>>>(h2all, W_ro, b_ro, out);
}
